// attentionBlock_73899207295044
// MI455X (gfx1250) — matrix operand order repaired and checked by reading, not compiled, not run on hardware
//
#include <hip/hip_runtime.h>
#include <hip/hip_bf16.h>

// ---------------------------------------------------------------------------
// Attention block for MI455X (gfx1250, wave32, WMMA, async-LDS copies).
//   B=32, S=1024, D=768, HD=64
//   Kernel 0: W[3][768,64] f32 -> W^T[3][64,768] f16 (one-time, tiny).
//   Kernel 1: QKV projection. x/W^T chunks staged to LDS with
//             global_load_async_to_lds_b128 (double-buffered, ASYNCcnt),
//             v_wmma_f32_16x16x32_f16 compute, Q/K f16 + V^T f16 out.
//   Kernel 2: flash attention. K/V^T tiles async-staged to LDS per block
//             (double-buffered), online softmax, WMMA both matmuls.
// ---------------------------------------------------------------------------

#define BB 32
#define SS 1024
#define DD 768
#define HD 64

typedef __attribute__((ext_vector_type(16))) _Float16 v16h;
typedef __attribute__((ext_vector_type(8)))  _Float16 v8h;
typedef __attribute__((ext_vector_type(4)))  float    v4f;
typedef __attribute__((ext_vector_type(8)))  float    v8f;

union Frag16 { v16h f; v8h h[2]; };

__device__ __forceinline__ v8f wmma_f16(v16h a, v16h b, v8f c) {
  return __builtin_amdgcn_wmma_f32_16x16x32_f16(false, a, false, b, (short)0, c,
                                                false, false);
}

__device__ __forceinline__ v8f v8f_zero() {
  v8f z = {0.f, 0.f, 0.f, 0.f, 0.f, 0.f, 0.f, 0.f};
  return z;
}

// Async global->LDS 16B copy (CDNA5, tracked by ASYNCcnt).
// The LDS destination is passed as a *pointer* and converted via ptrtoint:
// this captures the shared allocation, so the "memory" clobber legally
// covers it and the compiler cannot fold subsequent ds_loads to undef.
// (Generic pointers to LDS carry the LDS offset in their low 32 bits.)
__device__ __forceinline__ void async_copy_b128(void* lds_ptr, unsigned voff,
                                                const void* base) {
  const unsigned lds_off = (unsigned)(uintptr_t)lds_ptr;
  asm volatile("global_load_async_to_lds_b128 %0, %1, %2"
               :: "v"(lds_off), "v"(voff),
                  "s"((unsigned long long)(uintptr_t)base)
               : "memory");
}

__device__ __forceinline__ void wait_async0() {
  asm volatile("s_wait_asynccnt 0x0" ::: "memory");
}

// ---------------------------------------------------------------------------
// Kernel 0: W -> W^T f16.  wt[m][n*768 + k] = W_m[k*64 + n]
// ---------------------------------------------------------------------------
__global__ __launch_bounds__(256) void wconv_kernel(
    const float* __restrict__ Wq, const float* __restrict__ Wk,
    const float* __restrict__ Wv, _Float16* __restrict__ wt) {
  const int idx = blockIdx.x * 256 + threadIdx.x;     // [0, 3*768*64)
  const int m = idx / (DD * HD);
  const int rem = idx - m * (DD * HD);
  const int k = rem / HD;
  const int n = rem - k * HD;
  const float* W = (m == 0) ? Wq : ((m == 1) ? Wk : Wv);
  wt[(size_t)m * HD * DD + (size_t)n * DD + k] = (_Float16)W[rem];
}

// ---------------------------------------------------------------------------
// Kernel 1: QKV projection. grid = B*S/128, 256 threads (8 waves).
// LDS layout (56 KB, double buffered):
//   [0, 32768):       x chunk f32 [128 r][32 k], 16 KB per buffer
//   [32768, 57344):   W^T chunks f16 [3 m][64 n][32 k], 12 KB per buffer
// ---------------------------------------------------------------------------
#define X_BUF_BYTES 16384
#define W_BUF_BYTES 12288
#define W_BASE      (2 * X_BUF_BYTES)

__global__ __launch_bounds__(256) void qkv_proj_kernel(
    const float* __restrict__ x, const _Float16* __restrict__ wt,
    const float* __restrict__ bq, const float* __restrict__ bk,
    const float* __restrict__ bv,
    _Float16* __restrict__ q_out, _Float16* __restrict__ k_out,
    _Float16* __restrict__ vt_out) {
  __shared__ __align__(16) unsigned char smem[2 * X_BUF_BYTES + 2 * W_BUF_BYTES];

  const int tid = threadIdx.x;
  const int rowbase = blockIdx.x * 128;
  const int w = tid >> 5;
  const int lane = tid & 31;

  const float* xrowbase = x + (size_t)rowbase * DD;

  // Per-thread async-copy assignments (chunk geometry fixed; k0 varies).
  // x: 4 chunks of 16B each: cid -> r = cid>>3, ch = cid&7
  unsigned x_voff[4], x_lds[4];
#pragma unroll
  for (int i = 0; i < 4; ++i) {
    const int cid = tid + i * 256;
    const int r = cid >> 3, ch = cid & 7;
    x_voff[i] = (unsigned)r * (DD * 4) + (unsigned)ch * 16;   // + k0*4
    x_lds[i]  = (unsigned)r * 128 + (unsigned)ch * 16;        // + buf base
  }
  // W^T: 3 chunks: cid -> m = cid>>8, n = (cid&255)>>2, c = cid&3
  unsigned w_voff[3], w_lds[3];
#pragma unroll
  for (int i = 0; i < 3; ++i) {
    const int cid = tid + i * 256;
    const int m = cid >> 8, rem = cid & 255;
    const int n = rem >> 2, c = rem & 3;
    w_voff[i] = (unsigned)m * (HD * DD * 2) + (unsigned)n * (DD * 2) +
                (unsigned)c * 16;                             // + k0*2
    w_lds[i]  = (unsigned)m * 4096 + (unsigned)n * 64 +
                (unsigned)c * 16;                             // + buf base
  }

  auto issue_chunk = [&](int kc, int buf) {
    const unsigned k0 = (unsigned)kc * 32;
    unsigned char* xb = smem + buf * X_BUF_BYTES;
    unsigned char* wb = smem + W_BASE + buf * W_BUF_BYTES;
#pragma unroll
    for (int i = 0; i < 4; ++i)
      async_copy_b128(xb + x_lds[i], x_voff[i] + k0 * 4, xrowbase);
#pragma unroll
    for (int i = 0; i < 3; ++i)
      async_copy_b128(wb + w_lds[i], w_voff[i] + k0 * 2, wt);
  };

  v8f acc[3][4];
#pragma unroll
  for (int m = 0; m < 3; ++m)
#pragma unroll
    for (int nt = 0; nt < 4; ++nt) acc[m][nt] = v8f_zero();

  const int rl = w * 16 + (lane & 15);
  const int akb = (lane >> 4) * 8;
  const int kb2 = (lane >> 4) * 16;

  issue_chunk(0, 0);
  wait_async0();
  __syncthreads();

#pragma unroll 1
  for (int kc = 0; kc < DD / 32; ++kc) {
    const int cur = kc & 1;
    if (kc + 1 < DD / 32) issue_chunk(kc + 1, cur ^ 1);

    const float* xf = (const float*)(smem + cur * X_BUF_BYTES);
    const _Float16* wl =
        (const _Float16*)(smem + W_BASE + cur * W_BUF_BYTES);

    // Build A fragment (f32 LDS -> f16 regs). 16-bit A 16x32 layout.
    const float* xr = xf + rl * 32;
    const v4f x0 = *(const v4f*)(xr + akb);
    const v4f x1 = *(const v4f*)(xr + akb + 4);
    const v4f x2 = *(const v4f*)(xr + akb + 16);
    const v4f x3 = *(const v4f*)(xr + akb + 20);
    Frag16 a;
#pragma unroll
    for (int j = 0; j < 4; ++j) {
      a.f[j]      = (_Float16)x0[j];
      a.f[4 + j]  = (_Float16)x1[j];
      a.f[8 + j]  = (_Float16)x2[j];
      a.f[12 + j] = (_Float16)x3[j];
    }

#pragma unroll
    for (int m = 0; m < 3; ++m) {
      const _Float16* wm = wl + m * 2048;   // [n][32k] halves
      Frag16 bf[4];
#pragma unroll
      for (int nt = 0; nt < 4; ++nt) {
        const int nl = nt * 16 + (lane & 15);
        bf[nt].h[0] = *(const v8h*)(wm + nl * 32 + akb);
        bf[nt].h[1] = *(const v8h*)(wm + nl * 32 + akb + 16);
      }
#pragma unroll
      for (int nt = 0; nt < 4; ++nt)
        acc[m][nt] = wmma_f16(a.f, bf[nt].f, acc[m][nt]);
    }

    wait_async0();
    __syncthreads();
  }

  // Epilogue: bias + store. C layout: VGPR r -> row r + 8*(lane>>4).
  const int b = rowbase >> 10;
  const int s0 = (rowbase & (SS - 1)) + w * 16 + ((lane >> 4) << 3);

#pragma unroll
  for (int nt = 0; nt < 4; ++nt) {
    const int n = nt * 16 + (lane & 15);
    const float aq = bq[n], ak = bk[n], av = bv[n];
#pragma unroll
    for (int r = 0; r < 8; ++r) {
      const size_t sidx = (size_t)b * SS + s0 + r;
      q_out[sidx * HD + n] = (_Float16)(acc[0][nt][r] + aq);
      k_out[sidx * HD + n] = (_Float16)(acc[1][nt][r] + ak);
    }
    v8h pk;
#pragma unroll
    for (int r = 0; r < 8; ++r) pk[r] = (_Float16)(acc[2][nt][r] + av);
    *(v8h*)&vt_out[(size_t)b * HD * SS + (size_t)n * SS + s0] = pk;
  }
}

// ---------------------------------------------------------------------------
// Kernel 2: flash attention. grid = B*S/128, 256 threads (8 waves).
// LDS (24 KB): K tiles 2x4KB | V^T tiles 2x4KB | per-wave P staging 8KB.
//   K tile:  [32 key][64 hd] f16 (4KB, contiguous copy of K rows)
//   V tile:  [64 hd][32 key] f16 (4KB, strided rows of V^T)
// ---------------------------------------------------------------------------
#define KBUF 0
#define VBUF 8192
#define PBUF 16384

__global__ __launch_bounds__(256) void flash_attn_kernel(
    const _Float16* __restrict__ qh, const _Float16* __restrict__ kh,
    const _Float16* __restrict__ vt, float* __restrict__ out) {
  __shared__ __align__(16) unsigned char smem[PBUF + 8 * 1024];

  const int tid = threadIdx.x;
  const int b = blockIdx.x >> 3;
  const int qt = blockIdx.x & 7;
  const int w = tid >> 5;
  const int lane = tid & 31;

  const _Float16* kbat = kh + (size_t)b * SS * HD;
  const _Float16* vbat = vt + (size_t)b * HD * SS;

  // Q A-fragments held in registers for the whole kernel.
  const int qrow = qt * 128 + w * 16 + (lane & 15);
  const _Float16* qp = qh + ((size_t)b * SS + qrow) * HD;
  Frag16 aq[2];
#pragma unroll
  for (int ks = 0; ks < 2; ++ks) {
    const int hb = ks * 32 + ((lane >> 4) << 3);
    aq[ks].h[0] = *(const v8h*)(qp + hb);
    aq[ks].h[1] = *(const v8h*)(qp + hb + 16);
  }

  // Async tile staging assignments.
  const unsigned v_hd = (unsigned)(tid >> 2), v_c = (unsigned)(tid & 3);
  auto issue_tile = [&](int kt, int buf) {
    const unsigned key0 = (unsigned)kt * 32;
    // K tile: 4KB contiguous.
    async_copy_b128(smem + KBUF + buf * 4096 + tid * 16,
                    key0 * (HD * 2) + (unsigned)tid * 16, kbat);
    // V^T tile: 64 rows x 64B.
    async_copy_b128(smem + VBUF + buf * 4096 + v_hd * 64 + v_c * 16,
                    v_hd * (SS * 2) + key0 * 2 + v_c * 16, vbat);
  };

  float m_[8], l_[8];
  v8f acc[4];
#pragma unroll
  for (int r = 0; r < 8; ++r) { m_[r] = -1e30f; l_[r] = 0.f; }
#pragma unroll
  for (int nt = 0; nt < 4; ++nt) acc[nt] = v8f_zero();

  const float inv_d = 1.0f / (float)DD;
  const int prow = (lane >> 4) << 3;
  const int pcol = lane & 15;
  const int arow = lane & 15;
  const int akb = (lane >> 4) * 8;
  _Float16* pb = (_Float16*)(smem + PBUF + w * 1024);

  issue_tile(0, 0);
  wait_async0();
  __syncthreads();

#pragma unroll 1
  for (int kt = 0; kt < SS / 32; ++kt) {
    const int cur = kt & 1;
    if (kt + 1 < SS / 32) issue_tile(kt + 1, cur ^ 1);

    const _Float16* kb_l = (const _Float16*)(smem + KBUF + cur * 4096);
    const _Float16* vb_l = (const _Float16*)(smem + VBUF + cur * 4096);

    // ---- scores ----
    v8f sc[2];
#pragma unroll
    for (int j = 0; j < 2; ++j) {
      const int keyloc = j * 16 + (lane & 15);
      const _Float16* kp = kb_l + keyloc * 64;
      Frag16 bk0, bk1;
      bk0.h[0] = *(const v8h*)(kp + ((lane >> 4) << 3));
      bk0.h[1] = *(const v8h*)(kp + ((lane >> 4) << 3) + 16);
      bk1.h[0] = *(const v8h*)(kp + 32 + ((lane >> 4) << 3));
      bk1.h[1] = *(const v8h*)(kp + 32 + ((lane >> 4) << 3) + 16);
      v8f t = v8f_zero();
      t = wmma_f16(aq[0].f, bk0.f, t);
      t = wmma_f16(aq[1].f, bk1.f, t);
      sc[j] = t * inv_d;
    }

    // ---- online softmax stats ----
    float mnew[8], corr[8];
#pragma unroll
    for (int r = 0; r < 8; ++r) {
      float mx = fmaxf(sc[0][r], sc[1][r]);
#pragma unroll
      for (int off = 1; off < 16; off <<= 1)
        mx = fmaxf(mx, __shfl_xor(mx, off, 32));
      mnew[r] = fmaxf(m_[r], mx);
      corr[r] = __expf(m_[r] - mnew[r]);
      m_[r] = mnew[r];
    }

    // ---- P = exp(s - m) -> LDS (C-layout), row sums ----
#pragma unroll
    for (int r = 0; r < 8; ++r) {
      const float p0 = __expf(sc[0][r] - mnew[r]);
      const float p1 = __expf(sc[1][r] - mnew[r]);
      pb[(prow + r) * 32 + pcol]      = (_Float16)p0;
      pb[(prow + r) * 32 + 16 + pcol] = (_Float16)p1;
      float rs = p0 + p1;
#pragma unroll
      for (int off = 1; off < 16; off <<= 1) rs += __shfl_xor(rs, off, 32);
      l_[r] = l_[r] * corr[r] + rs;
    }

#pragma unroll
    for (int nt = 0; nt < 4; ++nt)
#pragma unroll
      for (int r = 0; r < 8; ++r) acc[nt][r] *= corr[r];

    asm volatile("s_wait_dscnt 0x0" ::: "memory");

    // ---- P back in A-layout, then acc += P @ V ----
    Frag16 ap;
    ap.h[0] = *(const v8h*)&pb[arow * 32 + akb];
    ap.h[1] = *(const v8h*)&pb[arow * 32 + akb + 16];

    Frag16 bv_[4];
#pragma unroll
    for (int nt = 0; nt < 4; ++nt) {
      const int n = nt * 16 + (lane & 15);
      const int kk = (lane >> 4) << 3;
      bv_[nt].h[0] = *(const v8h*)(vb_l + n * 32 + kk);
      bv_[nt].h[1] = *(const v8h*)(vb_l + n * 32 + kk + 16);
    }
#pragma unroll
    for (int nt = 0; nt < 4; ++nt)
      acc[nt] = wmma_f16(ap.f, bv_[nt].f, acc[nt]);

    wait_async0();
    __syncthreads();
  }

  // ---- epilogue: out = acc / l ----
  float* op = out + ((size_t)b * SS + qt * 128 + w * 16) * HD;
#pragma unroll
  for (int r = 0; r < 8; ++r) {
    const float inv = 1.0f / l_[r];
    const size_t row = (size_t)(r + ((lane >> 4) << 3));
#pragma unroll
    for (int nt = 0; nt < 4; ++nt) {
      const int n = nt * 16 + (lane & 15);
      op[row * HD + n] = acc[nt][r] * inv;
    }
  }
}

// ---------------------------------------------------------------------------
// Launch. Inputs (dict order): x, Wq, bq, Wk, bk, Wv, bv.
// Workspace: Qf16 | Kf16 | V^T f16 | W^T f16  (~12.9 MB).
// ---------------------------------------------------------------------------
extern "C" void kernel_launch(void* const* d_in, const int* in_sizes, int n_in,
                              void* d_out, int out_size, void* d_ws,
                              size_t ws_size, hipStream_t stream) {
  const float* x  = (const float*)d_in[0];
  const float* Wq = (const float*)d_in[1];
  const float* bq = (const float*)d_in[2];
  const float* Wk = (const float*)d_in[3];
  const float* bk = (const float*)d_in[4];
  const float* Wv = (const float*)d_in[5];
  const float* bv = (const float*)d_in[6];
  float* out = (float*)d_out;

  const size_t nqkv = (size_t)BB * SS * HD;
  _Float16* qh = (_Float16*)d_ws;
  _Float16* kh = qh + nqkv;
  _Float16* vt = kh + nqkv;
  _Float16* wt = vt + nqkv;

  wconv_kernel<<<(3 * DD * HD) / 256, 256, 0, stream>>>(Wq, Wk, Wv, wt);
  qkv_proj_kernel<<<(BB * SS) / 128, 256, 0, stream>>>(x, wt, bq, bk, bv,
                                                       qh, kh, vt);
  flash_attn_kernel<<<BB * (SS / 128), 256, 0, stream>>>(qh, kh, vt, out);
}
